// SparseCrossEncoderSelfAttention_37280316129300
// MI455X (gfx1250) — matrix operand order repaired and checked by reading, not compiled, not run on hardware
//
#include <hip/hip_runtime.h>
#include <hip/hip_bf16.h>

typedef __attribute__((ext_vector_type(16))) __bf16       v16bf;
typedef __attribute__((ext_vector_type(8)))  float        v8f;
typedef __attribute__((ext_vector_type(2)))  __bf16       bf16x2;
typedef __attribute__((ext_vector_type(4)))  unsigned int u32x4;

#define NUM_HEADS 12
#define HEAD_DIM  64
#define HIDDEN    768
#define B_        2
#define D_        16
#define Q_        32
#define S_        512
#define TOKB      8225            /* 1 + 32 + 16*512 tokens per batch */
#define NTOK      (B_ * TOKB)     /* 16450 */
#define NSEQ      545             /* 1 + 32 + 512 */
#define NKEYP     576             /* keys padded to 18*32 */
#define NQTILE    35              /* ceil(545/16) */
#define NJ        2304            /* 3*768 fused output dims */
#define SCALE_    0.125f          /* 1/sqrt(64) */

// ---------------------------------------------------------------- helpers

__device__ __forceinline__ __bf16 f2bf(float f) {
  unsigned u = __builtin_bit_cast(unsigned, f);
  u += 0x7FFFu + ((u >> 16) & 1u);          // round-to-nearest-even
  unsigned short h = (unsigned short)(u >> 16);
  return __builtin_bit_cast(__bf16, h);
}

__device__ __forceinline__ v8f wmma_bf16(v16bf a, v16bf b, v8f c) {
  return __builtin_amdgcn_wmma_f32_16x16x32_bf16(
      /*neg_a=*/false, a, /*neg_b=*/false, b,
      /*c_mod=*/(short)0, c, /*reuse_a=*/false, /*reuse_b=*/false);
}

__device__ __forceinline__ void ld2(v16bf& f, int i, const __bf16* p) {
  bf16x2 t = *(const bf16x2*)p;             // 4-byte aligned pair load
  f[2 * i]     = t.x;
  f[2 * i + 1] = t.y;
}

// token index within a batch for sequence position n (0..544) of doc d
__device__ __forceinline__ int keytok(int n, int d) {
  return (n < 33) ? n : (33 + d * S_ + (n - 33));
}

// ---------------------------------------------------------------- packing

__global__ void pack_x(const float* __restrict__ cls, const float* __restrict__ query,
                       const float* __restrict__ doc, __bf16* __restrict__ xb) {
  unsigned idx = blockIdx.x * 256u + threadIdx.x;
  const unsigned total = (unsigned)NTOK * HIDDEN;
  if (idx >= total) return;
  unsigned b = idx / ((unsigned)TOKB * HIDDEN);
  unsigned rem = idx % ((unsigned)TOKB * HIDDEN);
  unsigned t = rem / HIDDEN, k = rem % HIDDEN;
  float v;
  if (t == 0)       v = cls[b * HIDDEN + k];
  else if (t < 33)  v = query[(b * Q_ + (t - 1)) * HIDDEN + k];
  else              v = doc[((size_t)b * (D_ * S_) + (t - 33)) * HIDDEN + k];
  xb[idx] = f2bf(v);
}

__global__ void pack_w(const float* __restrict__ Wq, const float* __restrict__ Wk,
                       const float* __restrict__ Wv, __bf16* __restrict__ Wb) {
  unsigned idx = blockIdx.x * 256u + threadIdx.x;
  if (idx >= (unsigned)NJ * HIDDEN) return;
  unsigned j = idx / HIDDEN, k = idx % HIDDEN;
  float v = (j < 768)  ? Wq[j * HIDDEN + k]
          : (j < 1536) ? Wk[(j - 768) * HIDDEN + k]
                       : Wv[(j - 1536) * HIDDEN + k];
  Wb[idx] = f2bf(v);
}

__global__ void pack_b(const float* __restrict__ bq, const float* __restrict__ bk,
                       const float* __restrict__ bv, float* __restrict__ bb) {
  unsigned j = blockIdx.x * 256u + threadIdx.x;
  if (j >= (unsigned)NJ) return;
  bb[j] = (j < 768) ? bq[j] : (j < 1536) ? bk[j - 768] : bv[j - 1536];
}

// ---------------------------------------------------------------- QKV GEMM
// out[t, j] = sum_k xb[t,k] * Wb[j,k] + bb[j], split into Qp/Kp/Vp (bf16)

__global__ __launch_bounds__(256) void proj_gemm(
    const __bf16* __restrict__ xb, const __bf16* __restrict__ Wb,
    const float* __restrict__ bb,
    __bf16* __restrict__ Qp, __bf16* __restrict__ Kp, __bf16* __restrict__ Vp) {
  __shared__ __bf16 As[128][36];   // 128 tokens x 32 k (padded stride)
  __shared__ __bf16 Ws[64][36];    // 64 j x 32 k

  const int tid  = threadIdx.x;
  const int lane = tid & 31;
  const int wv   = tid >> 5;       // wave 0..7 -> rows wv*16..+15
  const int nn   = lane & 15;
  const int m0   = blockIdx.x * 128;
  const int n0   = blockIdx.y * 64;

  v8f acc[4];
#pragma unroll
  for (int j = 0; j < 4; ++j)
#pragma unroll
    for (int i = 0; i < 8; ++i) acc[j][i] = 0.0f;

  const int arow = tid >> 1, acol = (tid & 1) * 16;   // A staging map
  const int wrow = tid >> 2, wcol = (tid & 3) * 8;    // W staging map
  const int khB  = (lane >> 4) * 16;

  for (int kc = 0; kc < 24; ++kc) {
    {  // stage A tile (coalesced 128-bit loads)
      u32x4 d0 = {0, 0, 0, 0}, d1 = {0, 0, 0, 0};
      int gr = m0 + arow;
      if (gr < NTOK) {
        const u32x4* p = (const u32x4*)(xb + (size_t)gr * HIDDEN + kc * 32 + acol);
        d0 = p[0]; d1 = p[1];
      }
      unsigned int* dst = (unsigned int*)&As[arow][acol];
      dst[0] = d0.x; dst[1] = d0.y; dst[2] = d0.z; dst[3] = d0.w;
      dst[4] = d1.x; dst[5] = d1.y; dst[6] = d1.z; dst[7] = d1.w;
    }
    {  // stage W tile
      int gj = n0 + wrow;
      u32x4 dw = *(const u32x4*)(Wb + (size_t)gj * HIDDEN + kc * 32 + wcol);
      unsigned int* dst = (unsigned int*)&Ws[wrow][wcol];
      dst[0] = dw.x; dst[1] = dw.y; dst[2] = dw.z; dst[3] = dw.w;
    }
    __syncthreads();

    v16bf a;  // A fragment 16x32 (contiguous k = 16h + s)
    {
      const int mr = wv * 16 + nn;
      const int kh = (lane >> 4) * 16;
#pragma unroll
      for (int i = 0; i < 8; ++i) {
        int k = kh + 2 * i;
        ld2(a, i, &As[mr][k]);
      }
    }
#pragma unroll
    for (int jt = 0; jt < 4; ++jt) {
      v16bf bfr;  // B fragment 32x16: B[e][nn] = W[j0+nn][kc*32+e]
#pragma unroll
      for (int v = 0; v < 8; ++v) ld2(bfr, v, &Ws[jt * 16 + nn][khB + 2 * v]);
      acc[jt] = wmma_bf16(a, bfr, acc[jt]);
    }
    __syncthreads();
  }

  // epilogue: C layout row = i + 8*(lane>>4), col = nn
  const int rsel = (lane >> 4) * 8;
#pragma unroll
  for (int jt = 0; jt < 4; ++jt) {
    int jg = n0 + jt * 16 + nn;
    float bias = bb[jg];
    int mat = jg / HIDDEN;
    int jc  = jg % HIDDEN;
    __bf16* P = (mat == 0) ? Qp : ((mat == 1) ? Kp : Vp);
#pragma unroll
    for (int i = 0; i < 8; ++i) {
      int gr = m0 + wv * 16 + rsel + i;
      if (gr < NTOK) P[(size_t)gr * HIDDEN + jc] = f2bf(acc[jt][i] + bias);
    }
  }
}

// ---------------------------------------------------------------- attention
// one block per (b, d, h, q-group); 8 waves, one 16-row q-tile each;
// flash-style online softmax over 18 chunks of 32 keys.

__global__ __launch_bounds__(256) void attn(
    const __bf16* __restrict__ Qp, const __bf16* __restrict__ Kp,
    const __bf16* __restrict__ Vp, const float* __restrict__ qmask,
    const float* __restrict__ dmask, float* __restrict__ out) {
  __shared__ __bf16 Ks[32][68];      // 32 keys x 64 dims, row-major, padded
  __shared__ __bf16 Vts[64][34];     // V transposed: [dim][key], padded
  __shared__ float  Ms[NKEYP];       // additive mask, -1e30 in padding
  __shared__ __bf16 Ps[8][16][34];   // per-wave prob scratch (C->A transpose)

  const int tid  = threadIdx.x;
  const int lane = tid & 31;
  const int wv   = tid >> 5;
  const int nn   = lane & 15;
  const int qg   = blockIdx.x;
  const int h    = blockIdx.y;
  const int b    = blockIdx.z >> 4;
  const int d    = blockIdx.z & 15;

  for (int i = tid; i < NKEYP; i += 256) {
    float mval;
    if (i == 0)        mval = 0.0f;
    else if (i < 33)   mval = qmask[b * Q_ + (i - 1)];
    else if (i < NSEQ) mval = dmask[(b * D_ + d) * S_ + (i - 33)];
    else               mval = -1e30f;
    Ms[i] = mval;
  }

  const int  qtile  = qg * 8 + wv;
  const bool active = (qtile < NQTILE);
  const int  r0     = qtile * 16;

  v16bf a0, a1;  // Q fragments: dims 0..31 and 32..63
  {
    int qrow = r0 + nn;
    if (qrow > NSEQ - 1) qrow = NSEQ - 1;
    const __bf16* qb =
        Qp + ((size_t)(b * TOKB + keytok(qrow, d))) * HIDDEN + h * HEAD_DIM;
    const int kh = (lane >> 4) * 16;
#pragma unroll
    for (int i = 0; i < 8; ++i) {
      int k = kh + 2 * i;
      ld2(a0, i, qb + k);
      ld2(a1, i, qb + 32 + k);
    }
  }

  float mrow[8], lrow[8];
  v8f   o[4];
#pragma unroll
  for (int i = 0; i < 8; ++i) { mrow[i] = -1e30f; lrow[i] = 0.0f; }
#pragma unroll
  for (int j = 0; j < 4; ++j)
#pragma unroll
    for (int i = 0; i < 8; ++i) o[j][i] = 0.0f;

  const int kk  = tid >> 3;         // staging: key 0..31
  const int dp  = (tid & 7) * 8;    // staging: dim part
  const int khB = (lane >> 4) * 16;

  for (int kc = 0; kc < 18; ++kc) {
    __syncthreads();
    {  // cooperative stage of K rows + V transpose (zero pad past 545)
      int n = kc * 32 + kk;
      u32x4 kd = {0, 0, 0, 0}, vd = {0, 0, 0, 0};
      if (n < NSEQ) {
        size_t base =
            ((size_t)(b * TOKB + keytok(n, d))) * HIDDEN + h * HEAD_DIM + dp;
        kd = *(const u32x4*)(Kp + base);
        vd = *(const u32x4*)(Vp + base);
      }
      unsigned int* kr = (unsigned int*)&Ks[kk][dp];
      kr[0] = kd.x; kr[1] = kd.y; kr[2] = kd.z; kr[3] = kd.w;
      const __bf16* vb = (const __bf16*)&vd;
#pragma unroll
      for (int e = 0; e < 8; ++e) Vts[dp + e][kk] = vb[e];
    }
    __syncthreads();
    if (!active) continue;

    // scores: s[q][key] = sum_dim Q*K   (two 16x16 key tiles)
    v8f s_lo, s_hi;
#pragma unroll
    for (int i = 0; i < 8; ++i) { s_lo[i] = 0.0f; s_hi[i] = 0.0f; }
    {
      v16bf bA, bB;
#pragma unroll
      for (int v = 0; v < 8; ++v) {
        ld2(bA, v, &Ks[nn][khB + 2 * v]);
        ld2(bB, v, &Ks[nn][32 + khB + 2 * v]);
      }
      s_lo = wmma_bf16(a0, bA, s_lo);
      s_lo = wmma_bf16(a1, bB, s_lo);
#pragma unroll
      for (int v = 0; v < 8; ++v) {
        ld2(bA, v, &Ks[16 + nn][khB + 2 * v]);
        ld2(bB, v, &Ks[16 + nn][32 + khB + 2 * v]);
      }
      s_hi = wmma_bf16(a0, bA, s_hi);
      s_hi = wmma_bf16(a1, bB, s_hi);
    }
    const float mk_lo = Ms[kc * 32 + nn];
    const float mk_hi = Ms[kc * 32 + 16 + nn];
#pragma unroll
    for (int i = 0; i < 8; ++i) {
      s_lo[i] = s_lo[i] * SCALE_ + mk_lo;
      s_hi[i] = s_hi[i] * SCALE_ + mk_hi;
    }

    // online softmax per row (row i lives on 16 lanes of one half-wave)
#pragma unroll
    for (int i = 0; i < 8; ++i) {
      float cm = fmaxf(s_lo[i], s_hi[i]);
      cm = fmaxf(cm, __shfl_xor(cm, 1, 32));
      cm = fmaxf(cm, __shfl_xor(cm, 2, 32));
      cm = fmaxf(cm, __shfl_xor(cm, 4, 32));
      cm = fmaxf(cm, __shfl_xor(cm, 8, 32));
      float mn   = fmaxf(mrow[i], cm);
      float corr = __expf(mrow[i] - mn);
      float plo  = __expf(s_lo[i] - mn);
      float phi  = __expf(s_hi[i] - mn);
      float rs   = plo + phi;
      rs += __shfl_xor(rs, 1, 32);
      rs += __shfl_xor(rs, 2, 32);
      rs += __shfl_xor(rs, 4, 32);
      rs += __shfl_xor(rs, 8, 32);
      lrow[i] = lrow[i] * corr + rs;
      mrow[i] = mn;
#pragma unroll
      for (int j = 0; j < 4; ++j) o[j][i] *= corr;
      s_lo[i] = plo;
      s_hi[i] = phi;
    }

    // C-layout probs -> row-major LDS -> A fragment (per-wave scratch)
    const int rsel = (lane >> 4) * 8;
#pragma unroll
    for (int i = 0; i < 8; ++i) {
      Ps[wv][rsel + i][nn]      = f2bf(s_lo[i]);
      Ps[wv][rsel + i][16 + nn] = f2bf(s_hi[i]);
    }
    asm volatile("s_wait_dscnt 0" ::: "memory");  // same-wave DS RAW
    v16bf pa;
    {
      const int kh = (lane >> 4) * 16;
#pragma unroll
      for (int v = 0; v < 8; ++v) {
        int k = kh + 2 * v;
        ld2(pa, v, &Ps[wv][nn][k]);
      }
    }
    // ctx: o[q][dim] += P * V  (4 dim tiles)
#pragma unroll
    for (int j = 0; j < 4; ++j) {
      v16bf vfr;
#pragma unroll
      for (int v = 0; v < 8; ++v) ld2(vfr, v, &Vts[j * 16 + nn][khB + 2 * v]);
      o[j] = wmma_bf16(pa, vfr, o[j]);
    }
  }

  if (active) {
    const int rsel = (lane >> 4) * 8;
#pragma unroll
    for (int i = 0; i < 8; ++i) {
      int r = r0 + rsel + i;
      if (r < NSEQ) {
        float inv = 1.0f / lrow[i];
        size_t ob = (((size_t)(b * D_ + d)) * NSEQ + r) * HIDDEN + h * HEAD_DIM;
#pragma unroll
        for (int j = 0; j < 4; ++j) out[ob + j * 16 + nn] = o[j][i] * inv;
      }
    }
  }
}

// ---------------------------------------------------------------- launch

extern "C" void kernel_launch(void* const* d_in, const int* in_sizes, int n_in,
                              void* d_out, int out_size, void* d_ws, size_t ws_size,
                              hipStream_t stream) {
  (void)in_sizes; (void)n_in; (void)out_size; (void)ws_size;
  const float* cls   = (const float*)d_in[0];
  const float* query = (const float*)d_in[1];
  const float* doc   = (const float*)d_in[2];
  const float* qmask = (const float*)d_in[3];
  const float* dmask = (const float*)d_in[4];
  const float* Wq    = (const float*)d_in[5];
  const float* bq    = (const float*)d_in[6];
  const float* Wk    = (const float*)d_in[7];
  const float* bk    = (const float*)d_in[8];
  const float* Wv    = (const float*)d_in[9];
  const float* bv    = (const float*)d_in[10];
  float* out = (float*)d_out;

  char* ws = (char*)d_ws;
  size_t off = 0;
  auto carve = [&](size_t bytes) {
    char* p = ws + off;
    off = (off + bytes + 255) & ~(size_t)255;
    return p;
  };
  __bf16* xb = (__bf16*)carve((size_t)NTOK * HIDDEN * 2);
  __bf16* Wb = (__bf16*)carve((size_t)NJ * HIDDEN * 2);
  float*  bb = (float*)carve((size_t)NJ * 4);
  __bf16* Qp = (__bf16*)carve((size_t)NTOK * HIDDEN * 2);
  __bf16* Kp = (__bf16*)carve((size_t)NTOK * HIDDEN * 2);
  __bf16* Vp = (__bf16*)carve((size_t)NTOK * HIDDEN * 2);

  pack_x<<<((unsigned)NTOK * HIDDEN + 255) / 256, 256, 0, stream>>>(cls, query, doc, xb);
  pack_w<<<((unsigned)NJ * HIDDEN + 255) / 256, 256, 0, stream>>>(Wq, Wk, Wv, Wb);
  pack_b<<<(NJ + 255) / 256, 256, 0, stream>>>(bq, bk, bv, bb);
  proj_gemm<<<dim3((NTOK + 127) / 128, NJ / 64), 256, 0, stream>>>(xb, Wb, bb, Qp, Kp, Vp);
  attn<<<dim3(5, NUM_HEADS, B_ * D_), 256, 0, stream>>>(Qp, Kp, Vp, qmask, dmask, out);
}
